// MultiHeadSelfAttention_58772332479010
// MI455X (gfx1250) — matrix operand order repaired and checked by reading, not compiled, not run on hardware
//
#include <hip/hip_runtime.h>

// ---------------- problem constants ----------------
#define BATCH 4
#define SEQ   4096
#define CH    512
#define HEADS 8
#define DHEAD 64
#define RR    4
#define LRED  1024          // SEQ / RR
#define KSZ   7
#define PADW  3
#define EPSLN 1e-5f

typedef __bf16 bf16;
typedef __attribute__((ext_vector_type(16))) __bf16 bf16x16;
typedef __attribute__((ext_vector_type(8)))  __bf16 bf16x8;
typedef __attribute__((ext_vector_type(4)))  __bf16 bf16x4;
typedef __attribute__((ext_vector_type(8)))  float  f32x8;
typedef __attribute__((ext_vector_type(4)))  float  f32x4;
typedef __attribute__((ext_vector_type(2)))  int    i32x2;
typedef __attribute__((ext_vector_type(4)))  int    i32x4;

#define AS_STRIDE 40   // 32 + 8 halves pad: 80B row pitch keeps 16B alignment, dodges bank conflicts
#define BS_STRIDE 40

// ---------------- async global->LDS path (guarded; falls back to vector copies) ----------------
#if defined(__HIP_DEVICE_COMPILE__) && \
    __has_builtin(__builtin_amdgcn_global_load_async_to_lds_b128)
#define ASYNC_COPY 1
#else
#define ASYNC_COPY 0
#endif

#if ASYNC_COPY
typedef __attribute__((address_space(1))) i32x4 gvec4;
typedef __attribute__((address_space(3))) i32x4 lvec4;

__device__ __forceinline__ void async_cp_b128(void* lds, const void* g) {
    __attribute__((address_space(1))) void* gv =
        (__attribute__((address_space(1))) void*)const_cast<void*>(g);
    __attribute__((address_space(3))) void* lv =
        (__attribute__((address_space(3))) void*)lds;
    __builtin_amdgcn_global_load_async_to_lds_b128((gvec4*)gv, (lvec4*)lv, 0, 0);
}
#endif

template <int N>
__device__ __forceinline__ void wait_async() {
#if ASYNC_COPY
#if __has_builtin(__builtin_amdgcn_s_wait_asynccnt)
    __builtin_amdgcn_s_wait_asynccnt(N);
#else
    asm volatile("s_wait_asynccnt %0" :: "i"(N) : "memory");
#endif
#endif
}

// ---------------- f32 -> bf16 streaming convert (float4 -> bf16x4) ----------------
__global__ __launch_bounds__(256) void k_cvt_bf16(const float* __restrict__ src,
                                                  bf16* __restrict__ dst, int n) {
    const int i = (blockIdx.x * 256 + threadIdx.x) * 4;
    if (i < n) {
        f32x4 v = *(const f32x4*)(src + i);
        bf16x4 h;
        #pragma unroll
        for (int j = 0; j < 4; j++) h[j] = (bf16)v[j];
        *(bf16x4*)(dst + i) = h;
    }
}

// ---------------- conv1d reduce (stride RR, reflect pad) + LayerNorm -> bf16 ----------------
__global__ __launch_bounds__(256) void k_conv_red_ln(const float* __restrict__ x,
                                                     const float* __restrict__ w,
                                                     const float* __restrict__ gam,
                                                     const float* __restrict__ bet,
                                                     bf16* __restrict__ redb) {
    __shared__ __align__(16) float xt[KSZ][CH];
    __shared__ float sred[256];

    const int tid = threadIdx.x;
    const int bl  = blockIdx.x;          // 0 .. BATCH*LRED-1
    const int b   = bl / LRED;
    const int lr  = bl % LRED;

    // stage 7 reflect-padded input rows in LDS (float4 granularity)
    for (int i = tid; i < KSZ * (CH / 4); i += 256) {
        int k = i / (CH / 4), c4 = (i % (CH / 4)) * 4;
        int p = lr * RR - PADW + k;
        if (p < 0)    p = -p;
        if (p >= SEQ) p = 2 * SEQ - 2 - p;
        *(f32x4*)&xt[k][c4] = *(const f32x4*)&x[((size_t)b * SEQ + p) * CH + c4];
    }
    __syncthreads();

    // each thread computes 2 output channels
    float acc[2];
    #pragma unroll
    for (int t = 0; t < 2; t++) {
        const int co = tid + t * 256;
        const float* wr = &w[(size_t)co * CH * KSZ];
        float s = 0.f;
        for (int ci = 0; ci < CH; ci++) {
            #pragma unroll
            for (int k = 0; k < KSZ; k++) s += xt[k][ci] * wr[ci * KSZ + k];
        }
        acc[t] = s;
    }

    // block LayerNorm over the 512 channel values
    sred[tid] = acc[0] + acc[1];
    __syncthreads();
    for (int s = 128; s > 0; s >>= 1) { if (tid < s) sred[tid] += sred[tid + s]; __syncthreads(); }
    const float mean = sred[0] * (1.f / CH);
    __syncthreads();
    float d0 = acc[0] - mean, d1 = acc[1] - mean;
    sred[tid] = d0 * d0 + d1 * d1;
    __syncthreads();
    for (int s = 128; s > 0; s >>= 1) { if (tid < s) sred[tid] += sred[tid + s]; __syncthreads(); }
    const float inv = rsqrtf(sred[0] * (1.f / CH) + EPSLN);

    #pragma unroll
    for (int t = 0; t < 2; t++) {
        const int co = tid + t * 256;
        float o = (acc[t] - mean) * inv * gam[co] + bet[co];
        redb[((size_t)b * LRED + lr) * CH + co] = (bf16)o;
    }
}

// ---------------- tiled batched WMMA GEMM: C[M,N] = scale * (A[M,K] . B[N,K]^T) ----------------
// 256 threads = 8 wave32; block tile 64(M) x 64(N); waves 2x4, each owns a 32x16 C strip
// (two 16x16 accumulators sharing one B fragment -> 2 WMMAs per barrier pair).
// Double-buffered LDS k-tiles; async global->LDS copies when available.
// AF32: A operand is f32 in memory (converted to bf16 during staging).
template <bool AF32>
__global__ __launch_bounds__(256) void k_gemm_wmma(const void* __restrict__ Av,
                                                   const bf16* __restrict__ Bm,
                                                   float* __restrict__ Cm,
                                                   int M, int N, int K,
                                                   long aBatch, long bBatch, long cBatch,
                                                   float scale) {
    __shared__ __align__(16) bf16 As[2 * 64 * AS_STRIDE];
    __shared__ __align__(16) bf16 Bs[2 * 64 * BS_STRIDE];

    const int tid  = threadIdx.x;
    const int lane = tid & 31;
    const int wid  = tid >> 5;
    const int wm   = wid >> 2;      // 0..1  (32-row strip)
    const int wn   = wid & 3;       // 0..3  (16-col strip)

    const int mBase = blockIdx.y * 64;
    const int nBase = blockIdx.x * 64;

    const bf16*  Ab = AF32 ? nullptr : ((const bf16*)Av + (size_t)blockIdx.z * aBatch);
    const float* Af = AF32 ? ((const float*)Av + (size_t)blockIdx.z * aBatch) : nullptr;
    const bf16*  Bb = Bm + (size_t)blockIdx.z * bBatch;
    float*       Cp = Cm + (size_t)blockIdx.z * cBatch;

    // per-thread staging chunks: 64x32 tiles, 256 threads x 8 elems (b128)
    const int sR = tid >> 2;            // 0..63
    const int sJ = (tid & 3) * 8;

    const bf16*  aSrcH = AF32 ? nullptr : Ab + (size_t)(mBase + sR) * K + sJ;
    const float* aSrcF = AF32 ? Af + (size_t)(mBase + sR) * K + sJ : nullptr;
    const bf16*  bSrc  = Bb + (size_t)(nBase + sR) * K + sJ;
    bf16* aDst = &As[sR * AS_STRIDE + sJ];
    bf16* bDst = &Bs[sR * BS_STRIDE + sJ];

    auto stageTile = [&](int k0, int buf) {
        bf16* ad = aDst + buf * (64 * AS_STRIDE);
        bf16* bd = bDst + buf * (64 * BS_STRIDE);
        if (AF32) {
            f32x4 va0 = *(const f32x4*)(aSrcF + k0);
            f32x4 va1 = *(const f32x4*)(aSrcF + k0 + 4);
            __builtin_prefetch(aSrcF + k0 + 32, 0, 1);
            bf16x8 h;
            #pragma unroll
            for (int j = 0; j < 4; j++) { h[j] = (bf16)va0[j]; h[j + 4] = (bf16)va1[j]; }
            *(bf16x8*)ad = h;
        } else {
#if ASYNC_COPY
            async_cp_b128(ad, aSrcH + k0);
#else
            *(bf16x8*)ad = *(const bf16x8*)(aSrcH + k0);
#endif
        }
#if ASYNC_COPY
        async_cp_b128(bd, bSrc + k0);
#else
        *(bf16x8*)bd = *(const bf16x8*)(bSrc + k0);
#endif
    };

    f32x8 acc0 = {0.f, 0.f, 0.f, 0.f, 0.f, 0.f, 0.f, 0.f};
    f32x8 acc1 = {0.f, 0.f, 0.f, 0.f, 0.f, 0.f, 0.f, 0.f};

    // fragment addressing per ISA 16-bit A(16x32)/B(32x16) VGPR striping
    const int halfSel = lane >> 4;            // 0 or 1
    const int rowA0 = wm * 32 + (lane & 15);  // first 16-row subtile
    const int rowB  = wn * 16 + (lane & 15);
    const int kaLo = halfSel * 8;             // A low chunk K base
    const int kbLo = halfSel * 8;             // B low chunk K base

    const int nk = K >> 5;
    stageTile(0, 0);

    for (int kt = 0; kt < nk; ++kt) {
        const int buf = kt & 1;
        if (kt + 1 < nk) {
            stageTile((kt + 1) << 5, buf ^ 1);      // prefetch next tile into other buffer
            wait_async<(AF32 ? 1 : 2)>();           // in-order: current tile has landed
        } else {
            wait_async<0>();
        }
        __syncthreads();

        const bf16* asb = As + buf * (64 * AS_STRIDE);
        const bf16* bsb = Bs + buf * (64 * BS_STRIDE);

        // B fragment (shared by both accumulators):
        // lane<16 holds K {0..7,16..23}, lane>=16 holds K {8..15,24..31} of column (lane&15)
        bf16x8 b0 = *(const bf16x8*)&bsb[rowB * BS_STRIDE + kbLo];
        bf16x8 b1 = *(const bf16x8*)&bsb[rowB * BS_STRIDE + kbLo + 16];
        bf16x16 bfrag;
        #pragma unroll
        for (int i = 0; i < 8; i++) { bfrag[i] = b0[i]; bfrag[i + 8] = b1[i]; }

        // A fragments: lane<16 holds K {0..7,16..23}, lane>=16 holds K {8..15,24..31}
        bf16x8 a0 = *(const bf16x8*)&asb[rowA0 * AS_STRIDE + kaLo];
        bf16x8 a1 = *(const bf16x8*)&asb[rowA0 * AS_STRIDE + kaLo + 16];
        bf16x16 afrag0;
        #pragma unroll
        for (int i = 0; i < 8; i++) { afrag0[i] = a0[i]; afrag0[i + 8] = a1[i]; }

        bf16x8 a2 = *(const bf16x8*)&asb[(rowA0 + 16) * AS_STRIDE + kaLo];
        bf16x8 a3 = *(const bf16x8*)&asb[(rowA0 + 16) * AS_STRIDE + kaLo + 16];
        bf16x16 afrag1;
        #pragma unroll
        for (int i = 0; i < 8; i++) { afrag1[i] = a2[i]; afrag1[i + 8] = a3[i]; }

        acc0 = __builtin_amdgcn_wmma_f32_16x16x32_bf16(false, afrag0, false, bfrag,
                                                       (short)0, acc0, false, false);
        acc1 = __builtin_amdgcn_wmma_f32_16x16x32_bf16(false, afrag1, false, bfrag,
                                                       (short)0, acc1, false, false);
        __syncthreads();   // all waves done reading this buffer before it is restaged
    }

    // store: VGPR r, lane l -> row = r + 8*(l>>4), col = l&15
    const int colC = nBase + wn * 16 + (lane & 15);
    float* cp0 = Cp + (size_t)(mBase + wm * 32 + 8 * halfSel) * N + colC;
    float* cp1 = Cp + (size_t)(mBase + wm * 32 + 16 + 8 * halfSel) * N + colC;
    #pragma unroll
    for (int r = 0; r < 8; r++) {
        *cp0 = acc0[r] * scale; cp0 += N;
        *cp1 = acc1[r] * scale; cp1 += N;
    }
}

// ---------------- LayerNorm over C=512 + head-split to bf16 ----------------
// mode 0: out[(b,h,l,d)] (B,H,Lcur,DH) ; mode 1: out[(b,h,d,l)] (B,H,DH,Lcur) (pre-transposed V)
__global__ __launch_bounds__(256) void k_ln_split(const float* __restrict__ in,
                                                  const float* __restrict__ gam,
                                                  const float* __restrict__ bet,
                                                  bf16* __restrict__ outp,
                                                  int Lcur, int mode) {
    __shared__ float sred[256];
    const int tid = threadIdx.x;
    const int row = blockIdx.x;            // 0 .. BATCH*Lcur-1
    const int b = row / Lcur, l = row % Lcur;
    const float* rp = in + (size_t)row * CH;

    float v0 = rp[tid], v1 = rp[tid + 256];
    sred[tid] = v0 + v1;
    __syncthreads();
    for (int s = 128; s > 0; s >>= 1) { if (tid < s) sred[tid] += sred[tid + s]; __syncthreads(); }
    const float mean = sred[0] * (1.f / CH);
    __syncthreads();
    float d0 = v0 - mean, d1 = v1 - mean;
    sred[tid] = d0 * d0 + d1 * d1;
    __syncthreads();
    for (int s = 128; s > 0; s >>= 1) { if (tid < s) sred[tid] += sred[tid + s]; __syncthreads(); }
    const float inv = rsqrtf(sred[0] * (1.f / CH) + EPSLN);

    #pragma unroll
    for (int t = 0; t < 2; t++) {
        const int c = tid + t * 256;
        const float v = t ? v1 : v0;
        const float o = (v - mean) * inv * gam[c] + bet[c];
        const int h = c >> 6, d = c & 63;
        size_t idx = (mode == 0)
            ? (((size_t)(b * HEADS + h) * Lcur + l) * DHEAD + d)
            : (((size_t)(b * HEADS + h) * DHEAD + d) * Lcur + l);
        outp[idx] = (bf16)o;
    }
}

// ---------------- softmax in place over rows of length LRED ----------------
__global__ __launch_bounds__(256) void k_softmax(float* __restrict__ attn) {
    __shared__ float sred[256];
    const int tid = threadIdx.x;
    float* p = attn + (size_t)blockIdx.x * LRED;

    float v[4];
    #pragma unroll
    for (int i = 0; i < 4; i++) v[i] = p[tid + i * 256];
    float m = fmaxf(fmaxf(v[0], v[1]), fmaxf(v[2], v[3]));
    sred[tid] = m;
    __syncthreads();
    for (int s = 128; s > 0; s >>= 1) { if (tid < s) sred[tid] = fmaxf(sred[tid], sred[tid + s]); __syncthreads(); }
    m = sred[0];
    __syncthreads();
    float sum = 0.f;
    #pragma unroll
    for (int i = 0; i < 4; i++) { v[i] = __expf(v[i] - m); sum += v[i]; }
    sred[tid] = sum;
    __syncthreads();
    for (int s = 128; s > 0; s >>= 1) { if (tid < s) sred[tid] += sred[tid + s]; __syncthreads(); }
    const float inv = 1.f / sred[0];
    #pragma unroll
    for (int i = 0; i < 4; i++) p[tid + i * 256] = v[i] * inv;
}

// ---------------- (B,H,L,DH) f32 -> (B,L,C) bf16 head merge (float4 granularity) ----------------
__global__ __launch_bounds__(256) void k_merge_heads(const float* __restrict__ oh,
                                                     bf16* __restrict__ om) {
    const size_t q = ((size_t)blockIdx.x * 256 + threadIdx.x) * 4;   // merged (b,l,h,d) index
    const int d = (int)(q & 63);
    const size_t r = q >> 6;
    const int h = (int)(r & (HEADS - 1));
    const size_t r2 = r >> 3;
    const int l = (int)(r2 & (SEQ - 1));
    const int b = (int)(r2 >> 12);
    f32x4 v = *(const f32x4*)&oh[((size_t)(b * HEADS + h) * SEQ + l) * DHEAD + d];
    bf16x4 hv;
    #pragma unroll
    for (int j = 0; j < 4; j++) hv[j] = (bf16)v[j];
    *(bf16x4*)&om[q] = hv;
}

// ---------------- host orchestration ----------------
extern "C" void kernel_launch(void* const* d_in, const int* in_sizes, int n_in,
                              void* d_out, int out_size, void* d_ws, size_t ws_size,
                              hipStream_t stream) {
    const float* x   = (const float*)d_in[0];
    const float* Wq  = (const float*)d_in[1];
    const float* Wk  = (const float*)d_in[2];
    const float* Wv  = (const float*)d_in[3];
    const float* Wo  = (const float*)d_in[4];
    const float* cw  = (const float*)d_in[5];
    const float* g_r = (const float*)d_in[6];  const float* b_r = (const float*)d_in[7];
    const float* g_q = (const float*)d_in[8];  const float* b_q = (const float*)d_in[9];
    const float* g_k = (const float*)d_in[10]; const float* b_k = (const float*)d_in[11];
    const float* g_v = (const float*)d_in[12]; const float* b_v = (const float*)d_in[13];

    float* out_f = (float*)d_out;                               // (B,L,C)
    float* attn  = out_f + (size_t)BATCH * SEQ * CH;            // (B,H,L,LRED)

    char* wsb = (char*)d_ws;
    size_t off = 0;
    auto alloc = [&](size_t bytes) -> void* {
        void* p = (void*)(wsb + off);
        off += (bytes + 255) & ~(size_t)255;
        return p;
    };

    const size_t NX  = (size_t)BATCH * SEQ * CH;     // 8.39M
    const size_t NR  = (size_t)BATCH * LRED * CH;    // 2.10M
    const size_t NW  = (size_t)CH * CH;

    bf16* xb   = (bf16*)alloc(NX * 2);
    bf16* Wqb  = (bf16*)alloc(NW * 2);
    bf16* Wkb  = (bf16*)alloc(NW * 2);
    bf16* Wvb  = (bf16*)alloc(NW * 2);
    bf16* Wob  = (bf16*)alloc(NW * 2);
    bf16* redb = (bf16*)alloc(NR * 2);
    float* q_raw = (float*)alloc(NX * 4);
    float* k_raw = (float*)alloc(NR * 4);
    float* v_raw = (float*)alloc(NR * 4);
    bf16* qh = (bf16*)alloc(NX * 2);     // (B,H,SEQ,DH)
    bf16* kh = (bf16*)alloc(NR * 2);     // (B,H,LRED,DH)
    bf16* vT = (bf16*)alloc(NR * 2);     // (B,H,DH,LRED)
    float* out_heads = (float*)alloc(NX * 4);   // (B,H,SEQ,DH)
    bf16* out_merged = (bf16*)alloc(NX * 2);    // (B,SEQ,CH)

    // 1. bf16 copies of x and weights
    k_cvt_bf16<<<dim3((unsigned)(NX / 1024)), 256, 0, stream>>>(x, xb, (int)NX);
    k_cvt_bf16<<<dim3((unsigned)(NW / 1024)), 256, 0, stream>>>(Wq, Wqb, (int)NW);
    k_cvt_bf16<<<dim3((unsigned)(NW / 1024)), 256, 0, stream>>>(Wk, Wkb, (int)NW);
    k_cvt_bf16<<<dim3((unsigned)(NW / 1024)), 256, 0, stream>>>(Wv, Wvb, (int)NW);
    k_cvt_bf16<<<dim3((unsigned)(NW / 1024)), 256, 0, stream>>>(Wo, Wob, (int)NW);

    // 2. conv reduce + LN -> red bf16
    k_conv_red_ln<<<dim3(BATCH * LRED), 256, 0, stream>>>(x, cw, g_r, b_r, redb);

    // 3-5. projections (A . W^T)
    k_gemm_wmma<false><<<dim3(CH / 64, (BATCH * SEQ) / 64, 1), 256, 0, stream>>>(
        xb, Wqb, q_raw, BATCH * SEQ, CH, CH, 0, 0, 0, 1.f);
    k_gemm_wmma<false><<<dim3(CH / 64, (BATCH * LRED) / 64, 1), 256, 0, stream>>>(
        redb, Wkb, k_raw, BATCH * LRED, CH, CH, 0, 0, 0, 1.f);
    k_gemm_wmma<false><<<dim3(CH / 64, (BATCH * LRED) / 64, 1), 256, 0, stream>>>(
        redb, Wvb, v_raw, BATCH * LRED, CH, CH, 0, 0, 0, 1.f);

    // 6-8. LayerNorm + head split (V pre-transposed for the attn.V GEMM)
    k_ln_split<<<dim3(BATCH * SEQ),  256, 0, stream>>>(q_raw, g_q, b_q, qh, SEQ,  0);
    k_ln_split<<<dim3(BATCH * LRED), 256, 0, stream>>>(k_raw, g_k, b_k, kh, LRED, 0);
    k_ln_split<<<dim3(BATCH * LRED), 256, 0, stream>>>(v_raw, g_v, b_v, vT, LRED, 1);

    // 9. dots = qh . kh^T / sqrt(DH), batched over B*H, straight into d_out attn region
    k_gemm_wmma<false><<<dim3(LRED / 64, SEQ / 64, BATCH * HEADS), 256, 0, stream>>>(
        qh, kh, attn, SEQ, LRED, DHEAD,
        (long)SEQ * DHEAD, (long)LRED * DHEAD, (long)SEQ * LRED, 0.125f);

    // 10. softmax in place
    k_softmax<<<dim3(BATCH * HEADS * SEQ), 256, 0, stream>>>(attn);

    // 11. out_heads = attn . v  (A is f32, converted during staging; B = vT as (DH,LRED))
    k_gemm_wmma<true><<<dim3(DHEAD / 64, SEQ / 64, BATCH * HEADS), 256, 0, stream>>>(
        attn, vT, out_heads, SEQ, DHEAD, LRED,
        (long)SEQ * LRED, (long)DHEAD * LRED, (long)SEQ * DHEAD, 1.f);

    // 12. merge heads -> bf16
    k_merge_heads<<<dim3((unsigned)(NX / 1024)), 256, 0, stream>>>(out_heads, out_merged);

    // 13. final projection -> d_out out region
    k_gemm_wmma<false><<<dim3(CH / 64, (BATCH * SEQ) / 64, 1), 256, 0, stream>>>(
        out_merged, Wob, out_f, BATCH * SEQ, CH, CH, 0, 0, 0, 1.f);

    (void)in_sizes; (void)n_in; (void)out_size; (void)ws_size;
}
